// MultiHeadAttention_69595650064719
// MI455X (gfx1250) — matrix operand order repaired and checked by reading, not compiled, not run on hardware
//
#include <hip/hip_runtime.h>
#include <cstdint>
#include <cstddef>

// ---------------------------------------------------------------------------
// MI455X (gfx1250) fused multi-head attention, f16 WMMA with f32 accumulate.
// Round 3: async global->LDS builtin with the pointer types the compiler
// actually wants: pointer-to-(int __attribute__((vector_size(16)))) in AS1
// (global) / AS3 (LDS). Arity 4 confirmed by round-2 diagnostic.
// ---------------------------------------------------------------------------

typedef _Float16 v16h __attribute__((ext_vector_type(16)));
typedef _Float16 v8h  __attribute__((ext_vector_type(8)));
typedef _Float16 v4h  __attribute__((ext_vector_type(4)));
typedef float    v8f  __attribute__((ext_vector_type(8)));

#define WMMA_F32_F16(A, B, C) \
  __builtin_amdgcn_wmma_f32_16x16x32_f16(false, (A), false, (B), (short)0, (C), false, false)

// ---- async global->LDS (gfx1250) with safe fallback --------------------
#if defined(__has_builtin)
#if __has_builtin(__builtin_amdgcn_global_load_async_to_lds_b128)
#define HAVE_ASYNC_LDS 1
#endif
#endif
#ifndef HAVE_ASYNC_LDS
#define HAVE_ASYNC_LDS 0
#endif

#if HAVE_ASYNC_LDS
// exact pointee type from the round-2 diagnostic:
//   '__attribute__((__vector_size__(4 * sizeof(int)))) int __device__ *'
typedef int b128i __attribute__((vector_size(4 * sizeof(int))));
typedef __attribute__((address_space(1))) b128i* gbl_b128_ptr;
typedef __attribute__((address_space(3))) b128i* lds_b128_ptr;

__device__ __forceinline__ void async_copy_b128(const void* g, void* l) {
  __builtin_amdgcn_global_load_async_to_lds_b128(
      (gbl_b128_ptr)g, (lds_b128_ptr)l, /*offset=*/0, /*cpol=*/0);
}
__device__ __forceinline__ void async_wait0() {
#if __has_builtin(__builtin_amdgcn_s_wait_asynccnt)
  __builtin_amdgcn_s_wait_asynccnt(0);
#else
  asm volatile("s_wait_asynccnt 0x0" ::: "memory");
#endif
}
#endif

namespace cfg {
constexpr int EMB   = 1024;
constexpr int HEADS = 16;
constexpr int DH    = 64;      // head dim
constexpr int BATCH = 4;
constexpr int SEQ   = 1024;
constexpr int MROWS = BATCH * SEQ;  // 4096
constexpr int KDIM  = EMB;
constexpr int NDIM  = EMB;
// attention LDS strides (f16 elements), padded for conflict-free B-fragment reads
constexpr int KS_STRIDE = DH + 8;    // 72  (144B rows, 16B aligned)
constexpr int VS_STRIDE = SEQ + 8;   // 1032 (2064B rows, 16B aligned)
constexpr int PS_STRIDE = 40;        // 16x32 f16 P-tile, padded
}

// ---------------------------------------------------------------------------
// Projection GEMM:  Y[m,n] = sum_k A[m,k] * W[n,k] + bias[n]
//   A_HALF:  A is f16 (context) vs f32 (raw inputs, converted while staging)
//   OUT_QKV: write f16 into [B,H,S,D] layout vs f32 row-major [M,N]
// Block tile 128(M) x 64(N), 8 waves each computing 32x32, K-step 32.
// ---------------------------------------------------------------------------
template <bool A_HALF, bool OUT_QKV>
__global__ __launch_bounds__(256)
void proj_gemm(const void* __restrict__ Ap, const float* __restrict__ W,
               const float* __restrict__ bias, void* __restrict__ Outp) {
  using namespace cfg;
  __shared__ _Float16 As[128 * 40];  // A tile [128][32], stride 40 (80B rows)
  __shared__ _Float16 Bs[64 * 40];   // W tile [64][32] as Bs[n][k]

  const int tid  = threadIdx.x;
  const int lane = tid & 31;
  const int wid  = tid >> 5;
  const int m0   = blockIdx.y * 128;
  const int n0   = blockIdx.x * 64;
  const int wm   = wid >> 1;   // 0..3 : M sub-block of 32
  const int wn   = wid & 1;    // 0..1 : N sub-block of 32

  v8f acc[2][2] = {};

  for (int kb = 0; kb < KDIM; kb += 32) {
    // ---- stage A tile into LDS ----
    if (A_HALF) {
      const _Float16* A = (const _Float16*)Ap;
#pragma unroll
      for (int it = 0; it < 2; ++it) {
        const int idx = (tid + it * 256) * 8;
        const int row = idx >> 5, k = idx & 31;
#if HAVE_ASYNC_LDS
        // CDNA5 async DMA: global -> LDS without touching VGPRs
        async_copy_b128(A + (size_t)(m0 + row) * KDIM + kb + k,
                        &As[row * 40 + k]);
#else
        *(v8h*)&As[row * 40 + k] =
            *(const v8h*)(A + (size_t)(m0 + row) * KDIM + kb + k);
#endif
      }
    } else {
      const float* A = (const float*)Ap;
#pragma unroll
      for (int it = 0; it < 4; ++it) {
        const int idx = (tid + it * 256) * 4;
        const int row = idx >> 5, k = idx & 31;
        const float4 f = *(const float4*)(A + (size_t)(m0 + row) * KDIM + kb + k);
        v4h h = {(_Float16)f.x, (_Float16)f.y, (_Float16)f.z, (_Float16)f.w};
        *(v4h*)&As[row * 40 + k] = h;
      }
    }
    // ---- stage W tile (f32 -> f16), Bs[n][k] = W[n0+n][kb+k] ----
#pragma unroll
    for (int it = 0; it < 2; ++it) {
      const int idx = (tid + it * 256) * 4;
      const int n = idx >> 5, k = idx & 31;
      const float4 f = *(const float4*)(W + (size_t)(n0 + n) * KDIM + kb + k);
      v4h h = {(_Float16)f.x, (_Float16)f.y, (_Float16)f.z, (_Float16)f.w};
      *(v4h*)&Bs[n * 40 + k] = h;
    }
    // prefetch next W panel into cache (global_prefetch_b8 on gfx1250)
    if (kb + 32 < KDIM)
      __builtin_prefetch(W + (size_t)(n0 + (tid >> 2)) * KDIM + kb + 32, 0, 1);
#if HAVE_ASYNC_LDS
    if (A_HALF) async_wait0();   // drain ASYNCcnt before making LDS visible
#endif
    __syncthreads();

    // ---- fragments (per ISA 16-bit A / B VGPR layouts) ----
    const int kA = (lane >> 4) * 8;   // A frag: lane half selects K base
    const int kB = (lane >> 4) * 8;   // B frag: lane half selects K base
    v16h a[2], b[2];
#pragma unroll
    for (int tm = 0; tm < 2; ++tm) {
      const int row = wm * 32 + tm * 16 + (lane & 15);
      *(v8h*)&a[tm]       = *(const v8h*)&As[row * 40 + kA];        // K = kA..kA+7
      *((v8h*)&a[tm] + 1) = *(const v8h*)&As[row * 40 + kA + 16];   // K = kA+16..kA+23
    }
#pragma unroll
    for (int tn = 0; tn < 2; ++tn) {
      const int col = wn * 32 + tn * 16 + (lane & 15);
      *(v8h*)&b[tn]       = *(const v8h*)&Bs[col * 40 + kB];        // K = kB..kB+7
      *((v8h*)&b[tn] + 1) = *(const v8h*)&Bs[col * 40 + kB + 16];   // K = kB+16..kB+23
    }
#pragma unroll
    for (int tm = 0; tm < 2; ++tm)
#pragma unroll
      for (int tn = 0; tn < 2; ++tn)
        acc[tm][tn] = WMMA_F32_F16(a[tm], b[tn], acc[tm][tn]);
    __syncthreads();
  }

  // ---- epilogue: bias add + store ----
#pragma unroll
  for (int tm = 0; tm < 2; ++tm) {
#pragma unroll
    for (int tn = 0; tn < 2; ++tn) {
      const int n  = n0 + wn * 32 + tn * 16 + (lane & 15);
      const float bv = bias[n];
#pragma unroll
      for (int r = 0; r < 8; ++r) {
        const int m = m0 + wm * 32 + tm * 16 + r + 8 * (lane >> 4);
        const float val = acc[tm][tn][r] + bv;
        if (OUT_QKV) {
          const int bb = m >> 10, s = m & 1023;   // m = b*S + s
          const int h = n >> 6, d = n & 63;       // n = h*64 + d
          ((_Float16*)Outp)[(((size_t)(bb * HEADS + h)) * SEQ + s) * DH + d] =
              (_Float16)val;
        } else {
          ((float*)Outp)[(size_t)m * NDIM + n] = val;
        }
      }
    }
  }
}

// ---------------------------------------------------------------------------
// Fused attention for one (b,h) and 128 query rows per block (8 waves x 16 q).
// Full K and V head staged in LDS (CDNA5: 320KB/WGP). 3 passes over K tiles:
//   1) row max, 2) row exp-sum, 3) normalized weights -> HBM + P·V via WMMA.
// Scores exist only in WMMA accumulators — no score round-trip through HBM.
// ---------------------------------------------------------------------------
__global__ __launch_bounds__(256)
void attention_fused(const _Float16* __restrict__ Qh, const _Float16* __restrict__ Kh,
                     const _Float16* __restrict__ Vh, float* __restrict__ Wout,
                     _Float16* __restrict__ Ctx) {
  using namespace cfg;
  extern __shared__ char smem[];
  _Float16* Ks = (_Float16*)smem;                    // [1024][72]  row-major K
  _Float16* Vs = Ks + SEQ * KS_STRIDE;               // [64][1032]  V transposed
  _Float16* Ps = Vs + DH * VS_STRIDE;                // 8 x [16][40] P scratch

  const int tid  = threadIdx.x;
  const int lane = tid & 31;
  const int wave = tid >> 5;
  const int bh   = blockIdx.y;                       // b*HEADS + h
  const int q0   = blockIdx.x * 128 + wave * 16;

  const size_t base = (size_t)bh * SEQ * DH;
  const _Float16* Qp = Qh + base;
  const _Float16* Kp = Kh + base;
  const _Float16* Vp = Vh + base;

  // ---- stage K row-major (async DMA when available: pure byte copy) ----
  for (int it = 0; it < 32; ++it) {
    const int idx = (tid + it * 256) * 8;
    const int s = idx >> 6, d = idx & 63;
#if HAVE_ASYNC_LDS
    async_copy_b128(Kp + idx, &Ks[s * KS_STRIDE + d]);
#else
    *(v8h*)&Ks[s * KS_STRIDE + d] = *(const v8h*)(Kp + idx);
#endif
  }
  // ---- stage V transposed: Vs[d][s] (transforms data -> stays manual) ----
  for (int it = 0; it < 32; ++it) {
    const int idx = (tid + it * 256) * 8;
    const int s = idx >> 6, d = idx & 63;
    const v8h v = *(const v8h*)(Vp + idx);
#pragma unroll
    for (int j = 0; j < 8; ++j) Vs[(d + j) * VS_STRIDE + s] = v[j];
  }
#if HAVE_ASYNC_LDS
  async_wait0();
#endif
  __syncthreads();

  const int row = lane & 15;           // A/B fragment row-or-col index
  const int kA  = (lane >> 4) * 8;     // A fragment K base
  const int kB  = (lane >> 4) * 16;    // unused (B fragments use kA)

  // ---- Q fragments pinned in registers (covers d = 0..63) ----
  const _Float16* Qrow = Qp + (size_t)(q0 + row) * DH;
  v16h aQ0, aQ1;
  *(v8h*)&aQ0       = *(const v8h*)(Qrow + kA);
  *((v8h*)&aQ0 + 1) = *(const v8h*)(Qrow + kA + 16);
  *(v8h*)&aQ1       = *(const v8h*)(Qrow + 32 + kA);
  *((v8h*)&aQ1 + 1) = *(const v8h*)(Qrow + 32 + kA + 16);

  // 16x16 score tile at key columns [c0, c0+16): Q(16x64) x K^T via 2 WMMAs
  auto score_tile = [&](int c0) -> v8f {
    const _Float16* Kr = &Ks[(size_t)(c0 + row) * KS_STRIDE];
    v16h b0, b1;
    *(v8h*)&b0       = *(const v8h*)(Kr + kA);
    *((v8h*)&b0 + 1) = *(const v8h*)(Kr + kA + 16);
    *(v8h*)&b1       = *(const v8h*)(Kr + 32 + kA);
    *((v8h*)&b1 + 1) = *(const v8h*)(Kr + 32 + kA + 16);
    v8f z = {};
    v8f c = WMMA_F32_F16(aQ0, b0, z);
    c = WMMA_F32_F16(aQ1, b1, c);
    return c;
  };

  constexpr float SCALE = 0.125f;  // HEAD_DIM^-0.5
  float mrow[8], lrow[8];
#pragma unroll
  for (int r = 0; r < 8; ++r) { mrow[r] = -1e30f; lrow[r] = 0.0f; }

  // ---- pass 1: row max (per-lane partial, one cross-lane reduce) ----
  for (int kt = 0; kt < 64; ++kt) {
    const v8f sc = score_tile(kt * 16);
#pragma unroll
    for (int r = 0; r < 8; ++r) mrow[r] = fmaxf(mrow[r], sc[r] * SCALE);
  }
#pragma unroll
  for (int r = 0; r < 8; ++r)
#pragma unroll
    for (int msk = 1; msk < 16; msk <<= 1)
      mrow[r] = fmaxf(mrow[r], __shfl_xor(mrow[r], msk, 32));

  // ---- pass 2: row sum of exp ----
  for (int kt = 0; kt < 64; ++kt) {
    const v8f sc = score_tile(kt * 16);
#pragma unroll
    for (int r = 0; r < 8; ++r) lrow[r] += __expf(sc[r] * SCALE - mrow[r]);
  }
#pragma unroll
  for (int r = 0; r < 8; ++r) {
#pragma unroll
    for (int msk = 1; msk < 16; msk <<= 1)
      lrow[r] += __shfl_xor(lrow[r], msk, 32);
    lrow[r] = 1.0f / lrow[r];
  }

  // ---- pass 3: normalized weights out + context = P·V ----
  _Float16* Pw = Ps + wave * 16 * PS_STRIDE;   // per-wave scratch, no barrier needed
  float* wrow = Wout + (size_t)bh * SEQ * SEQ;
  v8f ctx[4] = {};
  for (int kt2 = 0; kt2 < 32; ++kt2) {
    const int c0 = kt2 * 32;
#pragma unroll
    for (int h2 = 0; h2 < 2; ++h2) {
      const v8f sc = score_tile(c0 + h2 * 16);
#pragma unroll
      for (int r = 0; r < 8; ++r) {
        const float p = __expf(sc[r] * SCALE - mrow[r]) * lrow[r];
        const int qr = r + 8 * (lane >> 4);
        wrow[(size_t)(q0 + qr) * SEQ + c0 + h2 * 16 + row] = p;  // weights (f32)
        Pw[qr * PS_STRIDE + h2 * 16 + row] = (_Float16)p;        // P tile (f16)
      }
    }
    // A fragment of P (16 q-rows x 32 key-cols), via per-wave LDS transpose
    v16h aP;
    *(v8h*)&aP       = *(const v8h*)&Pw[row * PS_STRIDE + kA];
    *((v8h*)&aP + 1) = *(const v8h*)&Pw[row * PS_STRIDE + kA + 16];
#pragma unroll
    for (int tn = 0; tn < 4; ++tn) {
      const _Float16* Vr = &Vs[(size_t)(tn * 16 + row) * VS_STRIDE + c0 + kA];
      v16h bV;
      *(v8h*)&bV       = *(const v8h*)Vr;
      *((v8h*)&bV + 1) = *(const v8h*)(Vr + 16);
      ctx[tn] = WMMA_F32_F16(aP, bV, ctx[tn]);
    }
  }

  // ---- epilogue: context -> [B,S,EMB] f16 workspace ----
  const int bb = bh >> 4, h = bh & 15;
#pragma unroll
  for (int tn = 0; tn < 4; ++tn)
#pragma unroll
    for (int r = 0; r < 8; ++r) {
      const int qr = q0 + r + 8 * (lane >> 4);
      const int d  = tn * 16 + row;
      Ctx[((size_t)(bb * SEQ + qr)) * EMB + h * DH + d] = (_Float16)ctx[tn][r];
    }
}

// ---------------------------------------------------------------------------
// Host launcher
// ---------------------------------------------------------------------------
extern "C" void kernel_launch(void* const* d_in, const int* in_sizes, int n_in,
                              void* d_out, int out_size, void* d_ws, size_t ws_size,
                              hipStream_t stream) {
  using namespace cfg;
  (void)in_sizes; (void)n_in; (void)out_size; (void)ws_size;

  const float* q  = (const float*)d_in[0];
  const float* k  = (const float*)d_in[1];
  const float* v  = (const float*)d_in[2];
  const float* Wq = (const float*)d_in[3];  const float* bq = (const float*)d_in[4];
  const float* Wk = (const float*)d_in[5];  const float* bk = (const float*)d_in[6];
  const float* Wv = (const float*)d_in[7];  const float* bv = (const float*)d_in[8];
  const float* Wo = (const float*)d_in[9];  const float* bo = (const float*)d_in[10];

  const size_t NE = (size_t)MROWS * EMB;  // 4M elements
  _Float16* qh  = (_Float16*)d_ws;        // [B,H,S,D] f16
  _Float16* kh  = qh + NE;
  _Float16* vh  = kh + NE;
  _Float16* ctx = vh + NE;                // [B,S,EMB] f16

  float* out = (float*)d_out;             // output [B,S,EMB] f32
  float* wts = out + NE;                  // weights [B,H,S,S] f32

  const dim3 gP(NDIM / 64, MROWS / 128);  // (16, 32)
  proj_gemm<false, true><<<gP, 256, 0, stream>>>(q, Wq, bq, qh);
  proj_gemm<false, true><<<gP, 256, 0, stream>>>(k, Wk, bk, kh);
  proj_gemm<false, true><<<gP, 256, 0, stream>>>(v, Wv, bv, vh);

  const size_t smem =
      (size_t)(SEQ * KS_STRIDE + DH * VS_STRIDE + 8 * 16 * PS_STRIDE) *
      sizeof(_Float16);  // ~283 KB, fits CDNA5's 320KB/WGP
  (void)hipFuncSetAttribute((const void*)attention_fused,
                            hipFuncAttributeMaxDynamicSharedMemorySize,
                            (int)smem);
  attention_fused<<<dim3(SEQ / 128, BATCH * HEADS), 256, smem, stream>>>(
      qh, kh, vh, wts, ctx);

  proj_gemm<true, false><<<gP, 256, 0, stream>>>(ctx, Wo, bo, out);
}
